// Transformer_10282151706707
// MI455X (gfx1250) — compile-verified, matrix operand order checked by reading, not run on hardware
//
#include <hip/hip_runtime.h>

// ---------------------------------------------------------------------------
// CDNA5 (gfx1250) transformer forward. All GEMMs via v_wmma_f32_16x16x32_f16,
// tile staging via GLOBAL_LOAD_ASYNC_TO_LDS_B128 (ASYNCcnt) where contiguous.
// ---------------------------------------------------------------------------

typedef __attribute__((ext_vector_type(16))) _Float16 v16h;
typedef __attribute__((ext_vector_type(8)))  float    v8f;
typedef _Float16 f16;

#define C_DIM 512
#define C4    2048
#define HD_   64
#define NH_   8
#define SEQ_  1024
#define NL_   4
#define NV_   32000
#define NB_   2

union Frag { v16h v; unsigned u[8]; };

// ---- CDNA5 async global->LDS staging helpers --------------------------------

__device__ __forceinline__ void async_load_b128(void* lds_ptr, const void* gptr) {
  // Per-lane: copies 16 bytes from global memory into LDS, tracked by ASYNCcnt.
  unsigned lds_off = (unsigned)(unsigned long long)lds_ptr;  // LDS aperture: low 32 bits
  asm volatile("global_load_async_to_lds_b128 %0, %1, off"
               :: "v"(lds_off), "v"(gptr) : "memory");
}

__device__ __forceinline__ void wait_asynccnt0() {
#if __has_builtin(__builtin_amdgcn_s_wait_asynccnt)
  __builtin_amdgcn_s_wait_asynccnt(0);
#else
  asm volatile("s_wait_asynccnt 0x0" ::: "memory");
#endif
}

// ---------------------------- elementwise helpers ---------------------------

__global__ __launch_bounds__(256) void k_conv_f16(const float* __restrict__ in,
                                                  f16* __restrict__ out, int n) {
  int i = blockIdx.x * 256 + threadIdx.x;
  if (i < n) out[i] = (f16)in[i];
}

// wq[L,H,C,HD] (f32) -> wqT[L, C, H*HD] (f16) so QKV proj is a plain GEMM.
__global__ __launch_bounds__(256) void k_qkvT(const float* __restrict__ in,
                                              f16* __restrict__ out, int n) {
  int i = blockIdx.x * 256 + threadIdx.x;
  if (i >= n) return;
  int d = i & 63;
  int c = (i >> 6) & 511;
  int h = (i >> 15) & 7;
  int l = i >> 18;
  out[((size_t)l * C_DIM + c) * C_DIM + h * HD_ + d] = (f16)in[i];
}

// x[r,c] = tok_emb[inp[r], c] + pos_emb[r % T, c]
__global__ __launch_bounds__(256) void k_embed(const int* __restrict__ inp,
                                               const float* __restrict__ tok,
                                               const float* __restrict__ pos,
                                               float* __restrict__ out, int total) {
  int i = blockIdx.x * 256 + threadIdx.x;
  if (i >= total) return;
  int c = i & 511;
  int r = i >> 9;
  int t = r & (SEQ_ - 1);
  out[i] = tok[(size_t)inp[r] * C_DIM + c] + pos[(size_t)t * C_DIM + c];
}

// -------------------------------- layernorm ---------------------------------
// one wave (32 lanes) per row of 512; writes f16 normalized output

__global__ __launch_bounds__(256) void k_layernorm(const float* __restrict__ x,
                                                   const float* __restrict__ g,
                                                   const float* __restrict__ b,
                                                   f16* __restrict__ out, int rows) {
  int wave = threadIdx.x >> 5, lane = threadIdx.x & 31;
  int row = blockIdx.x * 8 + wave;
  if (row >= rows) return;
  const float* xr = x + (size_t)row * C_DIM;
  float v[16];
  float s = 0.f;
#pragma unroll
  for (int i = 0; i < 16; ++i) { v[i] = xr[i * 32 + lane]; s += v[i]; }
#pragma unroll
  for (int o = 16; o > 0; o >>= 1) s += __shfl_xor(s, o, 32);
  float mean = s * (1.0f / C_DIM);
  float q = 0.f;
#pragma unroll
  for (int i = 0; i < 16; ++i) { float d = v[i] - mean; q += d * d; }
#pragma unroll
  for (int o = 16; o > 0; o >>= 1) q += __shfl_xor(q, o, 32);
  float rstd = rsqrtf(q * (1.0f / C_DIM) + 1e-5f);
  f16* orow = out + (size_t)row * C_DIM;
#pragma unroll
  for (int i = 0; i < 16; ++i) {
    int c = i * 32 + lane;
    orow[c] = (f16)((v[i] - mean) * rstd * g[c] + b[c]);
  }
}

// -------------------------------- softmax -----------------------------------
// one wave per row of S=1024 scores; scale=1/sqrt(64); causal and/or key mask;
// writes f16 probabilities for the P*V WMMA GEMM.

__global__ __launch_bounds__(256) void k_softmax(const float* __restrict__ s,
                                                 f16* __restrict__ p,
                                                 const int* __restrict__ mask,
                                                 int causal, int rows) {
  int wave = threadIdx.x >> 5, lane = threadIdx.x & 31;
  int row = blockIdx.x * 8 + wave;
  if (row >= rows) return;
  int t = row & (SEQ_ - 1);
  int z = row >> 10;           // (b*H + h)
  int b = z >> 3;              // z / NH_
  const float* sr = s + (size_t)row * SEQ_;
  float v[32];
  float mx = -3.4e38f;
#pragma unroll
  for (int i = 0; i < 32; ++i) {
    int col = i * 32 + lane;
    float val = sr[col] * 0.125f;               // HD^-0.5 = 1/8
    if (causal && col > t) val = -3.4e38f;
    if (mask && mask[b * SEQ_ + col] == 0) val = -3.4e38f;
    v[i] = val;
    mx = fmaxf(mx, val);
  }
#pragma unroll
  for (int o = 16; o > 0; o >>= 1) mx = fmaxf(mx, __shfl_xor(mx, o, 32));
  float sum = 0.f;
#pragma unroll
  for (int i = 0; i < 32; ++i) { v[i] = __expf(v[i] - mx); sum += v[i]; }
#pragma unroll
  for (int o = 16; o > 0; o >>= 1) sum += __shfl_xor(sum, o, 32);
  float inv = 1.0f / sum;
  f16* pr = p + (size_t)row * SEQ_;
#pragma unroll
  for (int i = 0; i < 32; ++i) pr[i * 32 + lane] = (f16)(v[i] * inv);
}

// ------------------------------ WMMA GEMM -----------------------------------
// C[z] = A[z](MxK) * B[z](KxN or NxK if transB) [+bias][+resid][relu]
// Block tile 128x64, 8 waves laid out 4(M) x 2(N), wave tile 32x32 = 2x2 WMMA.
// z = blockIdx.z; b = z/Hz, h = z%Hz with independent (b,h) strides per tensor.
// A tile and trans-B tile staged with GLOBAL_LOAD_ASYNC_TO_LDS_B128.

__global__ __launch_bounds__(256)
void k_gemm(const f16* __restrict__ A, int lda, long long sAb, long long sAh,
            const f16* __restrict__ B, int ldb, long long sBb, long long sBh,
            int transB,
            float* __restrict__ Cf, f16* __restrict__ Ch, int ldc,
            long long sCb, long long sCh,
            const float* __restrict__ bias, const float* __restrict__ resid,
            int relu, int M, int N, int K, int Hz) {
  __shared__ __align__(16) f16 As[128][40];   // [M][K] row-major, +8 pad
  __shared__ __align__(16) f16 Bs[64][40];    // [N][K] (B staged transposed)

  int tid  = threadIdx.x;
  int wave = tid >> 5, lane = tid & 31;
  int wm = wave >> 1, wn = wave & 1;
  int m0 = blockIdx.y * 128;
  int n0 = blockIdx.x * 64;
  int z  = blockIdx.z;
  int zb = z / Hz, zh = z % Hz;
  const f16* Ap = A + zb * sAb + zh * sAh;
  const f16* Bp = B + zb * sBb + zh * sBh;
  long long cbase = zb * sCb + zh * sCh;

  v8f zero = {0.f, 0.f, 0.f, 0.f, 0.f, 0.f, 0.f, 0.f};
  v8f acc[2][2];
#pragma unroll
  for (int i = 0; i < 2; ++i)
#pragma unroll
    for (int j = 0; j < 2; ++j) acc[i][j] = zero;

  int mrow  = lane & 15;   // A row / B col within a 16-tile
  int khalf = lane >> 4;   // which K half-group this lane holds

  for (int k0 = 0; k0 < K; k0 += 32) {
    // ---- stage A tile: 128x32 halves, 2 async b128 per thread, coalesced
#pragma unroll
    for (int i = 0; i < 2; ++i) {
      int e  = (tid + i * 256) * 8;
      int m  = e >> 5;
      int kk = e & 31;
      int mg = m0 + m; if (mg >= M) mg = M - 1;
      async_load_b128(&As[m][kk], Ap + (size_t)mg * lda + k0 + kk);
    }
    // ---- stage B tile transposed into Bs[n][k]
    if (transB) {           // B is [N,K]: contiguous K per lane -> async b128
      int e = tid * 8;
      int n = e >> 5, kk = e & 31;
      async_load_b128(&Bs[n][kk], Bp + (size_t)(n0 + n) * ldb + k0 + kk);
    } else {                // B is [K,N]: contiguous N reads, scatter into Bs
      int e  = tid * 8;
      int kk = e >> 6, n = e & 63;
      uint4 d = *(const uint4*)(Bp + (size_t)(k0 + kk) * ldb + n0 + n);
      f16 tmp[8];
      *(uint4*)tmp = d;
#pragma unroll
      for (int j = 0; j < 8; ++j) Bs[n + j][kk] = tmp[j];
    }
    if (k0 + 32 < K)  // gfx1250 global_prefetch_b8 of next A K-tile
      __builtin_prefetch(Ap + (size_t)(m0 + (tid >> 1)) * lda + k0 + 32, 0, 1);
    wait_asynccnt0();   // async LDS writes visible before barrier
    __syncthreads();

    // ---- load fragments per documented 16-bit layout:
    // element pair p holds k = 2p + (p>=4 ? 8 : 0) + khalf*8
    Frag af[2], bf[2];
#pragma unroll
    for (int t = 0; t < 2; ++t) {
      int m = wm * 32 + t * 16 + mrow;
      int n = wn * 32 + t * 16 + mrow;
#pragma unroll
      for (int pq = 0; pq < 8; ++pq) {
        int kk = 2 * pq + ((pq >= 4) ? 8 : 0) + khalf * 8;
        af[t].u[pq] = *(const unsigned*)&As[m][kk];
        bf[t].u[pq] = *(const unsigned*)&Bs[n][kk];
      }
    }
#pragma unroll
    for (int ti = 0; ti < 2; ++ti)
#pragma unroll
      for (int tj = 0; tj < 2; ++tj)
        acc[ti][tj] = __builtin_amdgcn_wmma_f32_16x16x32_f16(
            false, af[ti].v, false, bf[tj].v, (short)0, acc[ti][tj], false, false);
    __syncthreads();
  }

  // ---- epilogue: C/D layout -> lane n = lane&15, row m = r + (lane>>4)*8
  int nlo = lane & 15, mhi = lane >> 4;
#pragma unroll
  for (int ti = 0; ti < 2; ++ti)
#pragma unroll
    for (int tj = 0; tj < 2; ++tj) {
      int nn = n0 + wn * 32 + tj * 16 + nlo;
#pragma unroll
      for (int r = 0; r < 8; ++r) {
        int mm = m0 + wm * 32 + ti * 16 + mhi * 8 + r;
        if (mm < M && nn < N) {
          float val = acc[ti][tj][r];
          if (bias)  val += bias[nn];
          long long off = cbase + (long long)mm * ldc + nn;
          if (resid) val += resid[off];
          if (relu)  val = fmaxf(val, 0.f);
          if (Cf) Cf[off] = val;
          if (Ch) Ch[off] = (f16)val;
        }
      }
    }
}

// ------------------------------- host side ----------------------------------

static inline char* ws_take(char*& p, size_t bytes) {
  char* r = p;
  p += (bytes + 255) & ~(size_t)255;
  return r;
}

extern "C" void kernel_launch(void* const* d_in, const int* in_sizes, int n_in,
                              void* d_out, int out_size, void* d_ws, size_t ws_size,
                              hipStream_t stream) {
  (void)in_sizes; (void)n_in; (void)out_size; (void)ws_size;
  // ---- inputs in setup_inputs() order
  const int*   enc_inp  = (const int*)d_in[0];
  const int*   dec_inp  = (const int*)d_in[1];
  const int*   enc_mask = (const int*)d_in[2];
  const float* tok_emb  = (const float*)d_in[3];
  const float* pos_emb  = (const float*)d_in[4];
  const float* e_wq = (const float*)d_in[5],  *e_wk = (const float*)d_in[6];
  const float* e_wv = (const float*)d_in[7],  *e_wo = (const float*)d_in[8];
  const float* e_bo = (const float*)d_in[9];
  const float* sa_wq = (const float*)d_in[10], *sa_wk = (const float*)d_in[11];
  const float* sa_wv = (const float*)d_in[12], *sa_wo = (const float*)d_in[13];
  const float* sa_bo = (const float*)d_in[14];
  const float* ca_wq = (const float*)d_in[15], *ca_wk = (const float*)d_in[16];
  const float* ca_wv = (const float*)d_in[17], *ca_wo = (const float*)d_in[18];
  const float* ca_bo = (const float*)d_in[19];
  const float* e_ln1_g = (const float*)d_in[20], *e_ln1_b = (const float*)d_in[21];
  const float* e_ln2_g = (const float*)d_in[22], *e_ln2_b = (const float*)d_in[23];
  const float* d_ln1_g = (const float*)d_in[24], *d_ln1_b = (const float*)d_in[25];
  const float* d_ln2_g = (const float*)d_in[26], *d_ln2_b = (const float*)d_in[27];
  const float* d_ln3_g = (const float*)d_in[28], *d_ln3_b = (const float*)d_in[29];
  const float* e_w1 = (const float*)d_in[30], *e_b1 = (const float*)d_in[31];
  const float* e_w2 = (const float*)d_in[32], *e_b2 = (const float*)d_in[33];
  const float* d_w1 = (const float*)d_in[34], *d_b1 = (const float*)d_in[35];
  const float* d_w2 = (const float*)d_in[36], *d_b2 = (const float*)d_in[37];
  const float* out_w = (const float*)d_in[38], *out_b = (const float*)d_in[39];

  const int ROWS = NB_ * SEQ_;                      // 2048
  const size_t WCC  = (size_t)NL_ * C_DIM * C_DIM;  // per-stack qkv/wo f16 elems
  const size_t WC4  = (size_t)NL_ * C_DIM * C4;     // per-stack w1/w2 f16 elems
  const size_t SCOR = (size_t)NB_ * NH_ * SEQ_ * SEQ_;

  // ---- workspace carve-out
  char* p = (char*)d_ws;
  f16* wqT_e  = (f16*)ws_take(p, WCC * 2);
  f16* wkT_e  = (f16*)ws_take(p, WCC * 2);
  f16* wvT_e  = (f16*)ws_take(p, WCC * 2);
  f16* wqT_sa = (f16*)ws_take(p, WCC * 2);
  f16* wkT_sa = (f16*)ws_take(p, WCC * 2);
  f16* wvT_sa = (f16*)ws_take(p, WCC * 2);
  f16* wqT_ca = (f16*)ws_take(p, WCC * 2);
  f16* wkT_ca = (f16*)ws_take(p, WCC * 2);
  f16* wvT_ca = (f16*)ws_take(p, WCC * 2);
  f16* woh_e  = (f16*)ws_take(p, WCC * 2);
  f16* woh_sa = (f16*)ws_take(p, WCC * 2);
  f16* woh_ca = (f16*)ws_take(p, WCC * 2);
  f16* w1h_e  = (f16*)ws_take(p, WC4 * 2);
  f16* w2h_e  = (f16*)ws_take(p, WC4 * 2);
  f16* w1h_d  = (f16*)ws_take(p, WC4 * 2);
  f16* w2h_d  = (f16*)ws_take(p, WC4 * 2);
  f16* outwh  = (f16*)ws_take(p, (size_t)C_DIM * NV_ * 2);
  float* enc_x = (float*)ws_take(p, (size_t)ROWS * C_DIM * 4);
  float* dec_x = (float*)ws_take(p, (size_t)ROWS * C_DIM * 4);
  f16* hln   = (f16*)ws_take(p, (size_t)ROWS * C_DIM * 2);
  f16* hln2  = (f16*)ws_take(p, (size_t)ROWS * C_DIM * 2);
  f16* qb    = (f16*)ws_take(p, (size_t)ROWS * C_DIM * 2);
  f16* kb    = (f16*)ws_take(p, (size_t)ROWS * C_DIM * 2);
  f16* vb    = (f16*)ws_take(p, (size_t)ROWS * C_DIM * 2);
  f16* attnh = (f16*)ws_take(p, (size_t)ROWS * C_DIM * 2);
  f16* ffn1  = (f16*)ws_take(p, (size_t)ROWS * C4 * 2);
  float* scores = (float*)ws_take(p, SCOR * 4);
  f16* probs = (f16*)ws_take(p, SCOR * 2);

  auto cdiv = [](int a, int b) { return (a + b - 1) / b; };
  auto gemm = [&](const f16* A, int lda, long long sAb, long long sAh,
                  const f16* B, int ldb, long long sBb, long long sBh, int transB,
                  float* Cf, f16* Ch, int ldc, long long sCb, long long sCh,
                  const float* bias, const float* resid, int relu,
                  int M, int N, int K, int Hz, int Z) {
    dim3 grid(N / 64, (M + 127) / 128, Z);
    k_gemm<<<grid, 256, 0, stream>>>(A, lda, sAb, sAh, B, ldb, sBb, sBh, transB,
                                     Cf, Ch, ldc, sCb, sCh, bias, resid, relu,
                                     M, N, K, Hz);
  };

  // ---- weight precision conversion (f32 -> f16), deterministic each call
  int nqkv = NL_ * NH_ * C_DIM * HD_;
  k_qkvT<<<cdiv(nqkv, 256), 256, 0, stream>>>(e_wq, wqT_e, nqkv);
  k_qkvT<<<cdiv(nqkv, 256), 256, 0, stream>>>(e_wk, wkT_e, nqkv);
  k_qkvT<<<cdiv(nqkv, 256), 256, 0, stream>>>(e_wv, wvT_e, nqkv);
  k_qkvT<<<cdiv(nqkv, 256), 256, 0, stream>>>(sa_wq, wqT_sa, nqkv);
  k_qkvT<<<cdiv(nqkv, 256), 256, 0, stream>>>(sa_wk, wkT_sa, nqkv);
  k_qkvT<<<cdiv(nqkv, 256), 256, 0, stream>>>(sa_wv, wvT_sa, nqkv);
  k_qkvT<<<cdiv(nqkv, 256), 256, 0, stream>>>(ca_wq, wqT_ca, nqkv);
  k_qkvT<<<cdiv(nqkv, 256), 256, 0, stream>>>(ca_wk, wkT_ca, nqkv);
  k_qkvT<<<cdiv(nqkv, 256), 256, 0, stream>>>(ca_wv, wvT_ca, nqkv);
  k_conv_f16<<<cdiv((int)WCC, 256), 256, 0, stream>>>(e_wo, woh_e, (int)WCC);
  k_conv_f16<<<cdiv((int)WCC, 256), 256, 0, stream>>>(sa_wo, woh_sa, (int)WCC);
  k_conv_f16<<<cdiv((int)WCC, 256), 256, 0, stream>>>(ca_wo, woh_ca, (int)WCC);
  k_conv_f16<<<cdiv((int)WC4, 256), 256, 0, stream>>>(e_w1, w1h_e, (int)WC4);
  k_conv_f16<<<cdiv((int)WC4, 256), 256, 0, stream>>>(e_w2, w2h_e, (int)WC4);
  k_conv_f16<<<cdiv((int)WC4, 256), 256, 0, stream>>>(d_w1, w1h_d, (int)WC4);
  k_conv_f16<<<cdiv((int)WC4, 256), 256, 0, stream>>>(d_w2, w2h_d, (int)WC4);
  int now = C_DIM * NV_;
  k_conv_f16<<<cdiv(now, 256), 256, 0, stream>>>(out_w, outwh, now);

  // ---- embeddings
  int ntok = ROWS * C_DIM;
  k_embed<<<cdiv(ntok, 256), 256, 0, stream>>>(enc_inp, tok_emb, pos_emb, enc_x, ntok);
  k_embed<<<cdiv(ntok, 256), 256, 0, stream>>>(dec_inp, tok_emb, pos_emb, dec_x, ntok);

  const long long sXz = (long long)SEQ_ * C_DIM;  // per-batch stride in q/k/v
  const long long sSb = (long long)NH_ * SEQ_ * SEQ_, sSh = (long long)SEQ_ * SEQ_;
  const int ZHB = NB_ * NH_;
  const int smrows = ZHB * SEQ_;

  auto attention = [&](f16* hq, f16* hkv, const f16* wq, const f16* wk, const f16* wv,
                       const f16* wo, const float* bo, float* xres,
                       const int* msk, int causal, int l) {
    size_t wl = (size_t)l * C_DIM * C_DIM;
    gemm(hq, C_DIM, 0, 0, wq + wl, C_DIM, 0, 0, 0, nullptr, qb, C_DIM, 0, 0,
         nullptr, nullptr, 0, ROWS, C_DIM, C_DIM, 1, 1);
    gemm(hkv, C_DIM, 0, 0, wk + wl, C_DIM, 0, 0, 0, nullptr, kb, C_DIM, 0, 0,
         nullptr, nullptr, 0, ROWS, C_DIM, C_DIM, 1, 1);
    gemm(hkv, C_DIM, 0, 0, wv + wl, C_DIM, 0, 0, 0, nullptr, vb, C_DIM, 0, 0,
         nullptr, nullptr, 0, ROWS, C_DIM, C_DIM, 1, 1);
    // scores[b,h] = Q[b,h] @ K[b,h]^T
    gemm(qb, C_DIM, sXz, HD_, kb, C_DIM, sXz, HD_, 1,
         scores, nullptr, SEQ_, sSb, sSh, nullptr, nullptr, 0,
         SEQ_, SEQ_, HD_, NH_, ZHB);
    k_softmax<<<smrows / 8, 256, 0, stream>>>(scores, probs, msk, causal, smrows);
    // attn[b,h] = P @ V
    gemm(probs, SEQ_, sSb, sSh, vb, C_DIM, sXz, HD_, 0,
         nullptr, attnh, C_DIM, sXz, HD_, nullptr, nullptr, 0,
         SEQ_, HD_, SEQ_, NH_, ZHB);
    // x += attn @ wo + bo
    gemm(attnh, C_DIM, 0, 0, wo + wl, C_DIM, 0, 0, 0,
         xres, nullptr, C_DIM, 0, 0, bo + (size_t)l * C_DIM, xres, 0,
         ROWS, C_DIM, C_DIM, 1, 1);
  };
  auto ffn = [&](float* x, const float* lng, const float* lnb, const f16* w1,
                 const float* b1, const f16* w2, const float* b2, int l) {
    k_layernorm<<<ROWS / 8, 256, 0, stream>>>(x, lng + (size_t)l * C_DIM,
                                              lnb + (size_t)l * C_DIM, hln, ROWS);
    gemm(hln, C_DIM, 0, 0, w1 + (size_t)l * C_DIM * C4, C4, 0, 0, 0,
         nullptr, ffn1, C4, 0, 0, b1 + (size_t)l * C4, nullptr, 1,
         ROWS, C4, C_DIM, 1, 1);
    gemm(ffn1, C4, 0, 0, w2 + (size_t)l * C4 * C_DIM, C_DIM, 0, 0, 0,
         x, nullptr, C_DIM, 0, 0, b2 + (size_t)l * C_DIM, x, 0,
         ROWS, C_DIM, C4, 1, 1);
  };

  // ---- encoder
  for (int l = 0; l < NL_; ++l) {
    k_layernorm<<<ROWS / 8, 256, 0, stream>>>(enc_x, e_ln1_g + (size_t)l * C_DIM,
                                              e_ln1_b + (size_t)l * C_DIM, hln, ROWS);
    attention(hln, hln, wqT_e, wkT_e, wvT_e, woh_e, e_bo, enc_x, enc_mask, 0, l);
    ffn(enc_x, e_ln2_g, e_ln2_b, w1h_e, e_b1, w2h_e, e_b2, l);
  }

  // ---- decoder
  for (int l = 0; l < NL_; ++l) {
    // masked self-attention
    k_layernorm<<<ROWS / 8, 256, 0, stream>>>(dec_x, d_ln1_g + (size_t)l * C_DIM,
                                              d_ln1_b + (size_t)l * C_DIM, hln, ROWS);
    attention(hln, hln, wqT_sa, wkT_sa, wvT_sa, woh_sa, sa_bo, dec_x, nullptr, 1, l);
    // cross-attention: same LN applied to dec_x (queries) and enc_out (keys/vals)
    k_layernorm<<<ROWS / 8, 256, 0, stream>>>(dec_x, d_ln2_g + (size_t)l * C_DIM,
                                              d_ln2_b + (size_t)l * C_DIM, hln, ROWS);
    k_layernorm<<<ROWS / 8, 256, 0, stream>>>(enc_x, d_ln2_g + (size_t)l * C_DIM,
                                              d_ln2_b + (size_t)l * C_DIM, hln2, ROWS);
    attention(hln, hln2, wqT_ca, wkT_ca, wvT_ca, woh_ca, ca_bo, dec_x, enc_mask, 0, l);
    // FFN
    ffn(dec_x, d_ln3_g, d_ln3_b, w1h_d, d_b1, w2h_d, d_b2, l);
  }

  // ---- logits = dec_out @ out_w + out_b   (2048 x 32000 x 512)
  k_conv_f16<<<cdiv(ntok, 256), 256, 0, stream>>>(dec_x, hln, ntok);
  gemm(hln, C_DIM, 0, 0, outwh, NV_, 0, 0, 0,
       (float*)d_out, nullptr, NV_, 0, 0, out_b, nullptr, 0,
       ROWS, NV_, C_DIM, 1, 1);
}
